// ProductNeuralNetworkModel_86157043958291
// MI455X (gfx1250) — compile-verified, matrix operand order checked by reading, not run on hardware
//
#include <hip/hip_runtime.h>
#include <hip/hip_bf16.h>
#include <math.h>

// ---------------- model constants ----------------
#define NUM_FIELDS 16
#define FIELD_DIM  10000
#define EMBED_DIM  64
#define D0_        400
#define HID_       400
#define BATCH_     16384
#define K1_        1024          // NUM_FIELDS * EMBED_DIM
#define KPAD_      416           // 400 padded to multiple of 32 (13*32)
#define NPAD_      448           // 400 padded to multiple of 64 (7*64)
#define LDSROW     40            // LDS row stride in halves (80B: 16B aligned, conflict-free)

// 1/sqrt(1 + 1e-5)
#define BN_INV_STD 0.99999500003749969f

typedef __attribute__((ext_vector_type(16))) __bf16 v16bf;
typedef __attribute__((ext_vector_type(8)))  __bf16 v8bf;
typedef __attribute__((ext_vector_type(2)))  __bf16 v2bf;
typedef __attribute__((ext_vector_type(8)))  float  v8f;

// gfx1250 async global->LDS path (ASYNCcnt-tracked), if the toolchain exposes it
#if defined(__has_builtin)
#if __has_builtin(__builtin_amdgcn_global_load_async_to_lds_b128) && \
    __has_builtin(__builtin_amdgcn_s_wait_asynccnt)
#define USE_ASYNC_LDS 1
#endif
#endif

#ifdef USE_ASYNC_LDS
// builtin signature (from compiler diagnostic): param 1 is
// 'int __vector(4) __device__ *' (AS1), param 2 the LDS-side pointer (AS3)
typedef int async_v4i __attribute__((vector_size(16)));
#define GPTR(p) ((__attribute__((address_space(1))) async_v4i*)(p))
#define LPTR(p) ((__attribute__((address_space(3))) async_v4i*)(p))
#endif

// =====================================================================
// Kernel 1: embedding gather -> bf16 activation matrix A0 [BATCH][1024],
// plus FM quadratic term quad[b] = 0.5*((sum_f e)^2.sum - (e*e).sum)
// One wave (32 lanes) per batch row; each lane owns 2 embedding dims.
// =====================================================================
__global__ __launch_bounds__(256) void pnn_gather_quad(
    const int* __restrict__ x, const float* __restrict__ tbl,
    __bf16* __restrict__ A0, float* __restrict__ quad)
{
    const int gtid = blockIdx.x * blockDim.x + threadIdx.x;
    const int row  = gtid >> 5;
    const int lane = threadIdx.x & 31;
    if (row >= BATCH_) return;

    float s0 = 0.f, s1 = 0.f, e2 = 0.f;
#pragma unroll
    for (int f = 0; f < NUM_FIELDS; ++f) {
        const int idx = x[row * NUM_FIELDS + f] + f * FIELD_DIM;
        const float* src = tbl + (size_t)idx * EMBED_DIM + lane * 2;
        const float v0 = src[0];
        const float v1 = src[1];
        s0 += v0; s1 += v1;
        e2 += v0 * v0 + v1 * v1;
        v2bf p; p[0] = (__bf16)v0; p[1] = (__bf16)v1;
        *(v2bf*)(A0 + (size_t)row * K1_ + f * EMBED_DIM + lane * 2) = p;
    }
    float part = s0 * s0 + s1 * s1 - e2;
#pragma unroll
    for (int m = 16; m >= 1; m >>= 1) part += __shfl_xor(part, m, 32);
    if (lane == 0) quad[row] = 0.5f * part;
}

// =====================================================================
// Weight pre-pack: fp32 W[Nvalid][ldW] -> bf16 Wp[NPAD_][Kpad],
// zero-padded for n >= Nvalid or k >= Kvalid. Runs once, ~2MB traffic.
// =====================================================================
__global__ __launch_bounds__(256) void pnn_pack_w(
    const float* __restrict__ W, int ldW, int Nvalid, int Kvalid, int Kpad,
    __bf16* __restrict__ Wp, int total8)
{
    const int t = blockIdx.x * blockDim.x + threadIdx.x;
    if (t >= total8) return;
    const int k8 = (t * 8) % Kpad;
    const int n  = (t * 8) / Kpad;
    const float* wrow = W + (size_t)n * ldW;
    v8bf o;
#pragma unroll
    for (int j = 0; j < 8; ++j) {
        const int k = k8 + j;
        float v = 0.f;
        if (n < Nvalid && k < Kvalid) v = wrow[k];
        o[j] = (__bf16)v;
    }
    *(v8bf*)(Wp + (size_t)n * Kpad + k8) = o;
}

// =====================================================================
// Tiled bf16 WMMA GEMM: Out[b,n] = epilogue( sum_k A[b,k] * Wp[n,k] )
// Workgroup: 256 threads / 8 waves; tile 256 rows x 64 cols.
// Each wave: 32 rows x 64 cols = 8 accumulator tiles, K step 32.
// LDS double-buffered; refill of buffer cur^1 overlaps WMMA on cur,
// via async global->LDS copies (ASYNCcnt) when available, else via
// register staging.
// mode 0: val = acc + bias[n] + quad[b]
// mode 1: val = relu( gamma[n] * (acc + bias[n]) * BN_INV_STD + beta[n] )
// =====================================================================
__global__ __launch_bounds__(256) void pnn_gemm_bf16(
    const __bf16* __restrict__ A, int ldA,
    const __bf16* __restrict__ Wp, int Kpad,
    int Nvalid, __bf16* __restrict__ Out, int ldOut,
    int mode,
    const float* __restrict__ bias, const float* __restrict__ quad,
    const float* __restrict__ gamma, const float* __restrict__ beta)
{
    __shared__ __bf16 As[2][256 * LDSROW];   // 2 x 256 rows x 32 K   (40KB)
    __shared__ __bf16 Bs[2][64  * LDSROW];   // 2 x  64 cols x 32 K   (10KB)

    const int tid    = threadIdx.x;
    const int waveId = tid >> 5;
    const int lane   = tid & 31;
    const int lrow   = lane & 15;        // M (or N) index within tile
    const int lhi    = lane >> 4;        // K half-group select

    const int rowBlk = blockIdx.x * 256;
    const int colBlk = blockIdx.y * 64;

    const int bN    = tid >> 2;          // 0..63
    const int bPart = tid & 3;           // 0..3

    v8f acc[2][4];
#pragma unroll
    for (int rt = 0; rt < 2; ++rt)
#pragma unroll
        for (int ct = 0; ct < 4; ++ct) acc[rt][ct] = v8f(0.f);

#ifdef USE_ASYNC_LDS
    // ---- async stage: stream tile at k0 directly into LDS buffer `buf` ----
    auto async_stage = [&](int buf, int k0) {
#pragma unroll
        for (int i = 0; i < 4; ++i) {
            const int c    = tid + i * 256;
            const int r    = c >> 2;
            const int part = c & 3;
            __builtin_amdgcn_global_load_async_to_lds_b128(
                GPTR(A + (size_t)(rowBlk + r) * ldA + k0 + part * 8),
                LPTR(&As[buf][r * LDSROW + part * 8]), 0, 0);
        }
        __builtin_amdgcn_global_load_async_to_lds_b128(
            GPTR(Wp + (size_t)(colBlk + bN) * Kpad + k0 + bPart * 8),
            LPTR(&Bs[buf][bN * LDSROW + bPart * 8]), 0, 0);
    };

    async_stage(0, 0);
    __builtin_amdgcn_s_wait_asynccnt(0);
    __syncthreads();
#else
    v8bf aReg[4];
    v8bf bReg;
    auto load_regs = [&](int k0) {
#pragma unroll
        for (int i = 0; i < 4; ++i) {
            const int c    = tid + i * 256;
            const int r    = c >> 2;
            const int part = c & 3;
            aReg[i] = *(const v8bf*)(A + (size_t)(rowBlk + r) * ldA + k0 + part * 8);
        }
        bReg = *(const v8bf*)(Wp + (size_t)(colBlk + bN) * Kpad + k0 + bPart * 8);
    };
    auto store_lds = [&](int buf) {
#pragma unroll
        for (int i = 0; i < 4; ++i) {
            const int c    = tid + i * 256;
            const int r    = c >> 2;
            const int part = c & 3;
            *(v8bf*)&As[buf][r * LDSROW + part * 8] = aReg[i];
        }
        *(v8bf*)&Bs[buf][bN * LDSROW + bPart * 8] = bReg;
    };

    load_regs(0);
    store_lds(0);
    __syncthreads();
#endif

    int cur = 0;
    for (int k0 = 0; k0 < Kpad; k0 += 32) {
        const bool more = (k0 + 32 < Kpad);
        if (more) {
#ifdef USE_ASYNC_LDS
            async_stage(cur ^ 1, k0 + 32);                    // stream next tile into other buffer
#else
            load_regs(k0 + 32);                               // global loads in flight
#endif
            __builtin_prefetch(Wp + (size_t)(colBlk + bN) * Kpad + k0 + 64, 0, 1);
        }

        // ---- A fragments for this wave's two 16-row strips ----
        v16bf afrag[2];
#pragma unroll
        for (int rt = 0; rt < 2; ++rt) {
            const int abase = (waveId * 32 + rt * 16 + lrow) * LDSROW + lhi * 8;
            const v8bf a_lo = *(const v8bf*)&As[cur][abase];
            const v8bf a_hi = *(const v8bf*)&As[cur][abase + 16];
            afrag[rt] = __builtin_shufflevector(a_lo, a_hi,
                0,1,2,3,4,5,6,7, 8,9,10,11,12,13,14,15);
        }
        // ---- 4 column tiles: B fragment + 2 WMMAs each ----
#pragma unroll
        for (int ct = 0; ct < 4; ++ct) {
            const int bbase = (ct * 16 + lrow) * LDSROW + lhi * 8;
            const v8bf b_lo = *(const v8bf*)&Bs[cur][bbase];
            const v8bf b_hi = *(const v8bf*)&Bs[cur][bbase + 16];
            const v16bf bfrag = __builtin_shufflevector(b_lo, b_hi,
                0,1,2,3,4,5,6,7, 8,9,10,11,12,13,14,15);
            acc[0][ct] = __builtin_amdgcn_wmma_f32_16x16x32_bf16(
                false, afrag[0], false, bfrag, (short)0, acc[0][ct], false, false);
            acc[1][ct] = __builtin_amdgcn_wmma_f32_16x16x32_bf16(
                false, afrag[1], false, bfrag, (short)0, acc[1][ct], false, false);
        }

        if (more) {
#ifdef USE_ASYNC_LDS
            __builtin_amdgcn_s_wait_asynccnt(0);
#else
            store_lds(cur ^ 1);    // waits on the in-flight global loads
#endif
            __syncthreads();
            cur ^= 1;
        }
    }

    // ---- epilogue: C layout -> lanes 0-15: M=r, N=lane; lanes 16-31: M=r+8 ----
#pragma unroll
    for (int rt = 0; rt < 2; ++rt) {
        const int rowBase = rowBlk + waveId * 32 + rt * 16 + lhi * 8;
#pragma unroll
        for (int ct = 0; ct < 4; ++ct) {
            const int colg = colBlk + ct * 16 + lrow;
            if (colg >= ldOut) continue;
            const bool cvalid = (colg < Nvalid);
            const float bv = cvalid ? bias[colg] : 0.f;
            float gv = 0.f, bev = 0.f;
            if (mode == 1 && cvalid) { gv = gamma[colg]; bev = beta[colg]; }
#pragma unroll
            for (int r = 0; r < 8; ++r) {
                const int rowg = rowBase + r;
                float v = acc[rt][ct][r];
                if (mode == 0) {
                    v = v + bv + quad[rowg];
                } else {
                    v = gv * (v + bv) * BN_INV_STD + bev;
                    v = fmaxf(v, 0.f);
                }
                if (!cvalid) v = 0.f;
                Out[(size_t)rowg * ldOut + colg] = (__bf16)v;
            }
        }
    }
}

// =====================================================================
// Kernel 5: out[b] = sigmoid( dot(h3[b, 0:400], Wout) + bout )
// One wave per row.
// =====================================================================
__global__ __launch_bounds__(256) void pnn_head(
    const __bf16* __restrict__ A3, const float* __restrict__ Wout,
    const float* __restrict__ bout, float* __restrict__ out)
{
    const int gtid = blockIdx.x * blockDim.x + threadIdx.x;
    const int row  = gtid >> 5;
    const int lane = threadIdx.x & 31;
    if (row >= BATCH_) return;

    float p = 0.f;
    const __bf16* h = A3 + (size_t)row * KPAD_;
#pragma unroll
    for (int k = lane; k < HID_; k += 32) p += (float)h[k] * Wout[k];
#pragma unroll
    for (int m = 16; m >= 1; m >>= 1) p += __shfl_xor(p, m, 32);
    if (lane == 0) out[row] = 1.f / (1.f + expf(-(p + bout[0])));
}

// =====================================================================
// launch
// =====================================================================
extern "C" void kernel_launch(void* const* d_in, const int* in_sizes, int n_in,
                              void* d_out, int out_size, void* d_ws, size_t ws_size,
                              hipStream_t stream) {
    const int*   x     = (const int*)  d_in[0];
    const float* tbl   = (const float*)d_in[1];
    const float* lin_w = (const float*)d_in[2];   // [400][1024]
    const float* bias  = (const float*)d_in[3];
    const float* W1    = (const float*)d_in[4];   // [400][400]
    const float* b1    = (const float*)d_in[5];
    const float* g1    = (const float*)d_in[6];
    const float* be1   = (const float*)d_in[7];
    const float* W2    = (const float*)d_in[8];
    const float* b2    = (const float*)d_in[9];
    const float* g2    = (const float*)d_in[10];
    const float* be2   = (const float*)d_in[11];
    const float* Wout  = (const float*)d_in[12];
    const float* bout  = (const float*)d_in[13];
    float* out = (float*)d_out;

    // ---- workspace layout ----
    char* ws = (char*)d_ws;
    size_t off = 0;
    __bf16* A0   = (__bf16*)(ws + off); off += (size_t)BATCH_ * K1_ * 2;    // 32MB; reused as A2 [16384][416]
    float*  quad = (float*) (ws + off); off += (size_t)BATCH_ * 4;          // 64KB
    __bf16* B1   = (__bf16*)(ws + off); off += (size_t)BATCH_ * KPAD_ * 2;  // 13.3MB; A1 then A3
    __bf16* Wp1  = (__bf16*)(ws + off); off += (size_t)NPAD_ * K1_   * 2;   // 0.9MB
    __bf16* Wp2  = (__bf16*)(ws + off); off += (size_t)NPAD_ * KPAD_ * 2;   // 0.37MB
    __bf16* Wp3  = (__bf16*)(ws + off); off += (size_t)NPAD_ * KPAD_ * 2;   // 0.37MB

    // 0) pack weights to padded bf16 (once per launch, tiny)
    {
        const int t1 = NPAD_ * K1_   / 8;
        const int t2 = NPAD_ * KPAD_ / 8;
        pnn_pack_w<<<(t1 + 255) / 256, 256, 0, stream>>>(lin_w, K1_, D0_, K1_, K1_,   Wp1, t1);
        pnn_pack_w<<<(t2 + 255) / 256, 256, 0, stream>>>(W1,   D0_,  HID_, D0_, KPAD_, Wp2, t2);
        pnn_pack_w<<<(t2 + 255) / 256, 256, 0, stream>>>(W2,   HID_, HID_, HID_, KPAD_, Wp3, t2);
    }

    // 1) gather + quad
    pnn_gather_quad<<<(BATCH_ * 32) / 256, 256, 0, stream>>>(x, tbl, A0, quad);

    dim3 grid(BATCH_ / 256, NPAD_ / 64);   // (64, 7)

    // 2) lin = e @ lin_w^T + bias + quad  ->  A1 (bf16, [16384][416])
    pnn_gemm_bf16<<<grid, 256, 0, stream>>>(
        A0, K1_, Wp1, K1_, D0_, B1, KPAD_, 0, bias, quad, nullptr, nullptr);

    // 3) h = relu(bn(A1 @ W1^T + b1))  ->  A2 (reuses A0 region)
    pnn_gemm_bf16<<<grid, 256, 0, stream>>>(
        B1, KPAD_, Wp2, KPAD_, HID_, A0, KPAD_, 1, b1, nullptr, g1, be1);

    // 4) h = relu(bn(A2 @ W2^T + b2))  ->  A3 (reuses B1 region)
    pnn_gemm_bf16<<<grid, 256, 0, stream>>>(
        A0, KPAD_, Wp3, KPAD_, HID_, B1, KPAD_, 1, b2, nullptr, g2, be2);

    // 5) sigmoid head
    pnn_head<<<(BATCH_ * 32) / 256, 256, 0, stream>>>(B1, Wout, bout, out);

    (void)in_sizes; (void)n_in; (void)out_size; (void)ws_size;
}
